// NTXentLoss_89077621719714
// MI455X (gfx1250) — matrix operand order repaired and checked by reading, not compiled, not run on hardware
//
#include <hip/hip_runtime.h>

#define B_ROWS 4096
#define N_ROWS 8192
#define D_DIM  512
#define TEMP_INV  20.0f   // 1/0.05
#define TEMP_INV2 40.0f   // 1/(0.05*0.5)
#define KS 40             // padded LDS row stride (f16 elems): 20 dwords -> conflict-free
#define NTILE 64          // N_ROWS / 128
#define NBLOCKS_TILE 2080 // 64*65/2 upper-triangular tiles
#define NPAIRS 33550336.0f
#define NEGINF (-__builtin_inff())
#define ORD_NEG_INF 0x007FFFFFu

typedef __attribute__((ext_vector_type(8)))  _Float16 v8h;
typedef __attribute__((ext_vector_type(16))) _Float16 v16h;
typedef __attribute__((ext_vector_type(2)))  _Float16 v2h;
typedef __attribute__((ext_vector_type(8)))  float    v8f;

// monotonic float<->uint mapping so global atomicMax on u32 == float max (deterministic)
__device__ __forceinline__ unsigned f2ord(float f) {
  unsigned u = __float_as_uint(f);
  return (u & 0x80000000u) ? ~u : (u | 0x80000000u);
}
__device__ __forceinline__ float ord2f(unsigned x) {
  return (x & 0x80000000u) ? __uint_as_float(x & 0x7FFFFFFFu)
                           : __uint_as_float(~x);
}
// raw LDS byte offset: generic LDS address is {shared_aperture_hi32, offset}
__device__ __forceinline__ unsigned lds_off(const void* p) {
  return (unsigned)(uintptr_t)p;
}

// ---------------- K1: L2-normalize rows, emit f16 reps, exact f32 positives ----
__global__ void __launch_bounds__(256) k_normalize(
    const float* __restrict__ zi, const float* __restrict__ zj,
    _Float16* __restrict__ reps, float* __restrict__ pos,
    unsigned* __restrict__ rowmaxu)
{
  const int r = blockIdx.x;
  const int t = threadIdx.x;
  float2 a = ((const float2*)(zi + (size_t)r * D_DIM))[t];
  float2 b = ((const float2*)(zj + (size_t)r * D_DIM))[t];
  float ssi = a.x*a.x + a.y*a.y;
  float ssj = b.x*b.x + b.y*b.y;
  float dij = a.x*b.x + a.y*b.y;
  for (int m = 16; m; m >>= 1) {
    ssi += __shfl_xor(ssi, m, 32);
    ssj += __shfl_xor(ssj, m, 32);
    dij += __shfl_xor(dij, m, 32);
  }
  __shared__ float s0[8], s1[8], s2[8];
  int w = t >> 5, l = t & 31;
  if (l == 0) { s0[w] = ssi; s1[w] = ssj; s2[w] = dij; }
  __syncthreads();
  if (t == 0) {
    float x = 0.f, y = 0.f, z = 0.f;
    for (int i = 0; i < 8; ++i) { x += s0[i]; y += s1[i]; z += s2[i]; }
    s0[0] = x; s1[0] = y; s2[0] = z;
  }
  __syncthreads();
  float inv_i = 1.0f / fmaxf(sqrtf(s0[0]), 1e-12f);
  float inv_j = 1.0f / fmaxf(sqrtf(s1[0]), 1e-12f);
  if (t == 0) {
    float p = s2[0] * inv_i * inv_j;
    pos[r] = p; pos[r + B_ROWS] = p;
    rowmaxu[r] = ORD_NEG_INF; rowmaxu[r + B_ROWS] = ORD_NEG_INF;
  }
  v2h hi; hi.x = (_Float16)(a.x * inv_i); hi.y = (_Float16)(a.y * inv_i);
  v2h hj; hj.x = (_Float16)(b.x * inv_j); hj.y = (_Float16)(b.y * inv_j);
  ((v2h*)(reps + (size_t)r * D_DIM))[t] = hi;
  ((v2h*)(reps + (size_t)(r + B_ROWS) * D_DIM))[t] = hj;
}

// ---------------- K2: triangular-tiled WMMA GEMM with fused max / exp-sum -----
// 128 threads = 4 waves in a 2x2 grid; each wave owns a 64x64 subtile
// (16 accumulators) -> 16 WMMAs per 16 ds_load_b128 per K-slab.
__global__ void __launch_bounds__(128) k_simtile(
    const _Float16* __restrict__ reps,
    unsigned* __restrict__ rowmaxu,
    float* __restrict__ partial)
{
  __shared__ _Float16 As[2][128 * KS];
  __shared__ _Float16 Bs[2][128 * KS];
  __shared__ float maxbuf[128][2];
  __shared__ float colbuf[128][2];
  __shared__ float red[128];

  // decode upper-triangular tile (bi, bj>=bi)
  int tt = blockIdx.x, bi = 0;
  while (tt >= (NTILE - bi)) { tt -= (NTILE - bi); ++bi; }
  const int  bjt     = bi + tt;
  const bool diag    = (bjt == bi);
  const int  rowBase = bi  * 128;
  const int  colBase = bjt * 128;

  const int tid  = threadIdx.x;
  const int lane = tid & 31;
  const int wave = tid >> 5;
  const int wm   = wave >> 1;   // 0..1 : 64-row band
  const int wn   = wave & 1;    // 0..1 : 64-col band
  const int hs   = lane >> 4;   // half select
  const int l15  = lane & 15;

  v8f acc[4][4];
  const v8f vzero = {0.f,0.f,0.f,0.f,0.f,0.f,0.f,0.f};
  for (int i = 0; i < 4; ++i) for (int j = 0; j < 4; ++j) acc[i][j] = vzero;

  // issue async DMA of one 128x32 K-slab of A and B rows into LDS buffer `buf`
  // 512 16B-vectors per matrix, 128 threads -> 4 vectors each per matrix.
  auto issue_slab = [&](int kc, int buf) {
    #pragma unroll
    for (int q = 0; q < 4; ++q) {
      int vi  = tid + q * 128;
      int row = vi >> 2;
      int col = (vi & 3) << 3;
      unsigned aL = lds_off(&As[buf][row * KS + col]);
      unsigned bL = lds_off(&Bs[buf][row * KS + col]);
      unsigned ga = (unsigned)(((rowBase + row) * D_DIM + kc + col) * 2);
      unsigned gb = (unsigned)(((colBase + row) * D_DIM + kc + col) * 2);
      asm volatile("global_load_async_to_lds_b128 %0, %1, %2"
                   :: "v"(aL), "v"(ga), "s"(reps) : "memory");
      asm volatile("global_load_async_to_lds_b128 %0, %1, %2"
                   :: "v"(bL), "v"(gb), "s"(reps) : "memory");
    }
  };

  issue_slab(0, 0);

  int ib = 0;
  for (int kc = 0; kc < D_DIM; kc += 32, ib ^= 1) {
    __syncthreads();   // (A) all waves finished computing from the buffer we refill
    if (kc + 32 < D_DIM) {
      issue_slab(kc + 32, ib ^ 1);
      asm volatile("s_wait_asynccnt 0x8" ::: "memory");  // current slab landed
    } else {
      asm volatile("s_wait_asynccnt 0x0" ::: "memory");
    }
    __syncthreads();   // (B) current slab visible to all waves

    const _Float16* Ab = As[ib];
    const _Float16* Bb = Bs[ib];
    v16h afr[4], bfr[4];
    #pragma unroll
    for (int mt = 0; mt < 4; ++mt) {
      int row = wm * 64 + mt * 16 + l15;
      v8h lo = *(const v8h*)&Ab[row * KS + hs * 16];       // K 0..7 / 16..23
      v8h hi = *(const v8h*)&Ab[row * KS + hs * 16 + 8];   // K 8..15 / 24..31
      v16h a;
      #pragma unroll
      for (int e = 0; e < 8; ++e) { a[e] = lo[e]; a[e + 8] = hi[e]; }
      afr[mt] = a;
    }
    #pragma unroll
    for (int nt = 0; nt < 4; ++nt) {
      int row = wn * 64 + nt * 16 + l15;                   // column of sim = row of reps
      bfr[nt] = *(const v16h*)&Bb[row * KS + hs * 16];     // 16 contiguous K
    }
    #pragma unroll
    for (int mt = 0; mt < 4; ++mt)
      #pragma unroll
      for (int nt = 0; nt < 4; ++nt)
        acc[mt][nt] = __builtin_amdgcn_wmma_f32_16x16x32_f16(
            false, afr[mt], false, bfr[nt], (short)0, acc[mt][nt], false, false);
  }

  // -------- fused epilogue --------
  float esum = 0.0f;
  float cmax[4] = {NEGINF, NEGINF, NEGINF, NEGINF};  // this thread's 4 column maxes
  #pragma unroll
  for (int mt = 0; mt < 4; ++mt) {
    #pragma unroll
    for (int v = 0; v < 8; ++v) {
      const int lm = wm * 64 + mt * 16 + v + hs * 8;
      const int gi = rowBase + lm;
      float rmax = NEGINF;
      #pragma unroll
      for (int nt = 0; nt < 4; ++nt) {
        const int ln = wn * 64 + nt * 16 + l15;
        const int gj = colBase + ln;
        float s  = acc[mt][nt][v];
        float sm = (diag && (gi == gj)) ? NEGINF : s;
        rmax = fmaxf(rmax, sm);
        cmax[nt] = fmaxf(cmax[nt], sm);
        float wgt = (!diag || (gj > gi)) ? 1.0f : 0.0f;   // count each pair i<j once
        esum += wgt * __expf(4.0f * s - 4.0f);
      }
      for (int m = 8; m; m >>= 1) rmax = fmaxf(rmax, __shfl_xor(rmax, m, 32));
      if (l15 == 0) maxbuf[lm][wn] = rmax;
    }
  }
  // merge the two half-wave row groups' column maxes
  #pragma unroll
  for (int nt = 0; nt < 4; ++nt) {
    float c = fmaxf(cmax[nt], __shfl_xor(cmax[nt], 16, 32));
    if (hs == 0) colbuf[wn * 64 + nt * 16 + l15][wm] = c;
  }
  red[tid] = esum;
  __syncthreads();

  {
    float m = fmaxf(maxbuf[tid][0], maxbuf[tid][1]);
    atomicMax(rowmaxu + rowBase + tid, f2ord(m));          // order-independent
    if (!diag) {
      float c = fmaxf(colbuf[tid][0], colbuf[tid][1]);     // transpose contribution
      atomicMax(rowmaxu + colBase + tid, f2ord(c));
    }
  }
  for (int st = 64; st; st >>= 1) {
    if (tid < st) red[tid] += red[tid + st];
    __syncthreads();
  }
  if (tid == 0) partial[blockIdx.x] = red[0];
}

// ---------------- K3: finalize CE + uniformity ---------------------------------
__global__ void __launch_bounds__(256) k_finalize(
    const unsigned* __restrict__ rowmaxu,
    const float* __restrict__ pos,
    const float* __restrict__ partial,
    float* __restrict__ out)
{
  const int tid = threadIdx.x;
  float ce = 0.0f, us = 0.0f;
  for (int i = tid; i < N_ROWS; i += 256) {
    float hn = ord2f(rowmaxu[i]);
    float l0 = pos[i] * TEMP_INV;
    float l1 = hn * TEMP_INV2;
    float mm = fmaxf(l0, l1);
    ce += logf(expf(l0 - mm) + expf(l1 - mm)) + mm - l0;   // -log_softmax[0]
  }
  for (int b = tid; b < NBLOCKS_TILE; b += 256) us += partial[b];
  __shared__ float r0[256], r1[256];
  r0[tid] = ce; r1[tid] = us;
  __syncthreads();
  for (int st = 128; st; st >>= 1) {
    if (tid < st) { r0[tid] += r0[tid + st]; r1[tid] += r1[tid + st]; }
    __syncthreads();
  }
  if (tid == 0) out[0] = r0[0] / (float)N_ROWS + 0.2f * logf(r1[0] / NPAIRS);
}

extern "C" void kernel_launch(void* const* d_in, const int* in_sizes, int n_in,
                              void* d_out, int out_size, void* d_ws, size_t ws_size,
                              hipStream_t stream)
{
  const float* zi = (const float*)d_in[0];
  const float* zj = (const float*)d_in[1];
  char* ws = (char*)d_ws;
  const size_t reps_bytes = (size_t)N_ROWS * D_DIM * sizeof(_Float16); // 8 MB
  _Float16* reps    = (_Float16*)ws;
  unsigned* rowmaxu = (unsigned*)(ws + reps_bytes);
  float*    pos     = (float*)(ws + reps_bytes + (size_t)N_ROWS * 4);
  float*    partial = (float*)(ws + reps_bytes + (size_t)N_ROWS * 8);

  k_normalize<<<B_ROWS, 256, 0, stream>>>(zi, zj, reps, pos, rowmaxu);
  k_simtile<<<NBLOCKS_TILE, 128, 0, stream>>>(reps, rowmaxu, partial);
  k_finalize<<<1, 256, 0, stream>>>(rowmaxu, pos, partial, (float*)d_out);
}
